// CTGRU_Model_26276609917071
// MI455X (gfx1250) — matrix operand order repaired and checked by reading, not compiled, not run on hardware
//
#include <hip/hip_runtime.h>

typedef __attribute__((ext_vector_type(16))) _Float16 v16h;
typedef __attribute__((ext_vector_type(8)))  _Float16 v8h;
typedef __attribute__((ext_vector_type(8)))  float    v8f;

#define LN10_HALF 1.15129254649702295f   // 0.5 * ln(10)

union HFrag { v16h v; v8h h[2]; };

// ---------------- utility kernels ----------------

__global__ __launch_bounds__(256) void zero_f32(float* __restrict__ p, int n) {
  int i = blockIdx.x * blockDim.x + threadIdx.x;
  if (i < n) p[i] = 0.0f;
}

// Wt[n*K + k] = (f16) W[k*N + n]  (transpose so each output-column's K run is contiguous)
__global__ __launch_bounds__(256) void conv_transpose_f16(const float* __restrict__ W,
                                                          _Float16* __restrict__ Wt,
                                                          int K, int N) {
  int i = blockIdx.x * blockDim.x + threadIdx.x;
  if (i >= K * N) return;
  int n = i / K;
  int k = i - n * K;
  Wt[i] = (_Float16)W[(size_t)k * N + n];
}

// fused[b, 0:256] = x[b, t, :]; fused[b, 256:768] = h (= hbuf, prev hidden sum)
// also seed qin's x-part (its h-part is written after retrieval)
__global__ __launch_bounds__(256) void build_fused(const float* __restrict__ x,
                                                   const float* __restrict__ hbuf,
                                                   _Float16* __restrict__ fused,
                                                   _Float16* __restrict__ qin, int t) {
  int i = blockIdx.x * blockDim.x + threadIdx.x;
  if (i >= 4096 * 768) return;
  int b = i / 768;
  int c = i - b * 768;
  float v = (c < 256) ? x[((size_t)b * 16 + t) * 256 + c]
                      : hbuf[(size_t)b * 512 + (c - 256)];
  _Float16 hv = (_Float16)v;
  fused[i] = hv;
  if (c < 256) qin[i] = hv;
}

// ---------------- WMMA GEMM: C[M,N] = A[M,K] @ Bt[N,K]^T + bias (opt tanh) ----------------
// Block: 256 threads = 8 waves; block tile 128x128; wave tile 32x64 (2x4 WMMAs).
// K loop is software-pipelined with two fragment buffers (manually peeled double-step).
template <bool TANH>
__global__ __launch_bounds__(256) void gemm_f16_wmma(const _Float16* __restrict__ A,
                                                     const _Float16* __restrict__ Bt,
                                                     const float* __restrict__ bias,
                                                     float* __restrict__ C,
                                                     int Ndim, int Kdim) {
  const int lane   = threadIdx.x & 31;
  const int wave   = threadIdx.x >> 5;
  const int laneLo = lane & 15;
  const int hi     = lane >> 4;                 // 0 or 1 (half-wave select)
  const int rowBase = blockIdx.y * 128 + (wave >> 1) * 32;
  const int colBase = blockIdx.x * 128 + (wave & 1) * 64;

  // Per-fragment base pointers (lane-resolved), advanced by K offsets only.
  const _Float16* pA[2];
  const _Float16* pB[4];
#pragma unroll
  for (int mi = 0; mi < 2; ++mi)
    pA[mi] = A + (size_t)(rowBase + mi * 16 + laneLo) * Kdim + hi * 16;
#pragma unroll
  for (int ni = 0; ni < 4; ++ni)
    pB[ni] = Bt + (size_t)(colBase + ni * 16 + laneLo) * Kdim + hi * 16;

  v8f acc[2][4] = {};

  HFrag a0[2], a1[2];
  v16h  b0[4], b1[4];

  // Prologue: prefetch K-tiles 0 and 32 into the two buffers.
#pragma unroll
  for (int mi = 0; mi < 2; ++mi) {
    a0[mi].h[0] = *(const v8h*)(pA[mi]);
    a0[mi].h[1] = *(const v8h*)(pA[mi] + 8);
    a1[mi].h[0] = *(const v8h*)(pA[mi] + 32);
    a1[mi].h[1] = *(const v8h*)(pA[mi] + 40);
  }
#pragma unroll
  for (int ni = 0; ni < 4; ++ni) {
    b0[ni] = *(const v16h*)(pB[ni]);
    b1[ni] = *(const v16h*)(pB[ni] + 32);
  }

  // Main loop: consume buf0/buf1, reload each right after use (k+64 / k+96).
  int k = 0;
  for (; k + 64 < Kdim; k += 64) {
#pragma unroll
    for (int mi = 0; mi < 2; ++mi)
#pragma unroll
      for (int ni = 0; ni < 4; ++ni)
        acc[mi][ni] = __builtin_amdgcn_wmma_f32_16x16x32_f16(
            false, a0[mi].v, false, b0[ni], (short)0, acc[mi][ni], false, false);
#pragma unroll
    for (int mi = 0; mi < 2; ++mi) {
      a0[mi].h[0] = *(const v8h*)(pA[mi] + k + 64);
      a0[mi].h[1] = *(const v8h*)(pA[mi] + k + 72);
    }
#pragma unroll
    for (int ni = 0; ni < 4; ++ni)
      b0[ni] = *(const v16h*)(pB[ni] + k + 64);

#pragma unroll
    for (int mi = 0; mi < 2; ++mi)
#pragma unroll
      for (int ni = 0; ni < 4; ++ni)
        acc[mi][ni] = __builtin_amdgcn_wmma_f32_16x16x32_f16(
            false, a1[mi].v, false, b1[ni], (short)0, acc[mi][ni], false, false);
#pragma unroll
    for (int mi = 0; mi < 2; ++mi) {
      a1[mi].h[0] = *(const v8h*)(pA[mi] + k + 96);
      a1[mi].h[1] = *(const v8h*)(pA[mi] + k + 104);
    }
#pragma unroll
    for (int ni = 0; ni < 4; ++ni)
      b1[ni] = *(const v16h*)(pB[ni] + k + 96);
  }

  // Epilogue: last two K-tiles (already loaded).
#pragma unroll
  for (int mi = 0; mi < 2; ++mi)
#pragma unroll
    for (int ni = 0; ni < 4; ++ni)
      acc[mi][ni] = __builtin_amdgcn_wmma_f32_16x16x32_f16(
          false, a0[mi].v, false, b0[ni], (short)0, acc[mi][ni], false, false);
#pragma unroll
  for (int mi = 0; mi < 2; ++mi)
#pragma unroll
    for (int ni = 0; ni < 4; ++ni)
      acc[mi][ni] = __builtin_amdgcn_wmma_f32_16x16x32_f16(
          false, a1[mi].v, false, b1[ni], (short)0, acc[mi][ni], false, false);

  // Epilogue store: C layout — lane = column (laneLo), VGPR r = row (hi*8 + r)
#pragma unroll
  for (int mi = 0; mi < 2; ++mi)
#pragma unroll
    for (int ni = 0; ni < 4; ++ni) {
      const int col  = colBase + ni * 16 + laneLo;
      const float bn = bias[col];
      const int row0 = rowBase + mi * 16 + hi * 8;
      float* cp = C + (size_t)row0 * Ndim + col;
#pragma unroll
      for (int r = 0; r < 8; ++r) {
        float v = acc[mi][ni][r] + bn;
        if (TANH) v = tanhf(v);
        cp[(size_t)r * Ndim] = v;
      }
    }
}

// ---------------- softmax(r) + retrieval: rh = sum_j r_j * h_hat_j -> qin h-part ----------------
__global__ __launch_bounds__(256) void softmax_r_qin(const float* __restrict__ R,
                                                     const float* __restrict__ hhat,
                                                     _Float16* __restrict__ qin) {
  int i = blockIdx.x * blockDim.x + threadIdx.x;
  if (i >= 4096 * 512) return;
  const float* z  = R    + (size_t)i * 8;
  const float* hh = hhat + (size_t)i * 8;
  float v[8], m = -1e30f;
#pragma unroll
  for (int j = 0; j < 8; ++j) {
    float d = z[j] - (float)j * LN10_HALF;
    v[j] = -d * d;
    m = fmaxf(m, v[j]);
  }
  float s = 0.0f;
#pragma unroll
  for (int j = 0; j < 8; ++j) { v[j] = __expf(v[j] - m); s += v[j]; }
  const float inv = 1.0f / s;
  float rh = 0.0f;
#pragma unroll
  for (int j = 0; j < 8; ++j) rh += v[j] * inv * hh[j];
  int b = i >> 9, u = i & 511;
  qin[(size_t)b * 768 + 256 + u] = (_Float16)rh;
}

// ---------------- softmax(s) + state update + hidden sum ----------------
__global__ __launch_bounds__(256) void update_step(const float* __restrict__ S,
                                                   const float* __restrict__ qbuf,
                                                   float* __restrict__ hhat,
                                                   float* __restrict__ hbuf) {
  int i = blockIdx.x * blockDim.x + threadIdx.x;
  if (i >= 4096 * 512) return;
  const float* z = S + (size_t)i * 8;
  float* hh = hhat + (size_t)i * 8;
  float v[8], m = -1e30f;
#pragma unroll
  for (int j = 0; j < 8; ++j) {
    float d = z[j] - (float)j * LN10_HALF;
    v[j] = -d * d;
    m = fmaxf(m, v[j]);
  }
  float s = 0.0f;
#pragma unroll
  for (int j = 0; j < 8; ++j) { v[j] = __expf(v[j] - m); s += v[j]; }
  const float inv = 1.0f / s;
  const float q = qbuf[i];
  float hs = 0.0f;
#pragma unroll
  for (int j = 0; j < 8; ++j) {
    float sk  = v[j] * inv;
    float dec = __expf(-0.04f / ((float)j * LN10_HALF + 1e-7f)); // j=0 -> 0 (underflow), matches ref
    float hn  = ((1.0f - sk) * hh[j] + sk * q) * dec;
    hh[j] = hn;
    hs += hn;
  }
  hbuf[i] = hs;
}

// ---------------- out[b,t,:] = hbuf[b,:] @ W_out + b_out ----------------
__global__ __launch_bounds__(64) void out_proj(const float* __restrict__ hbuf,
                                               const float* __restrict__ Wo,
                                               const float* __restrict__ bo,
                                               float* __restrict__ out, int t) {
  __shared__ float sh[64 * 3];
  const int b = blockIdx.x, tid = threadIdx.x;
  float p0 = 0.f, p1 = 0.f, p2 = 0.f;
  for (int u = tid; u < 512; u += 64) {
    float h = hbuf[(size_t)b * 512 + u];
    p0 += h * Wo[u * 3 + 0];
    p1 += h * Wo[u * 3 + 1];
    p2 += h * Wo[u * 3 + 2];
  }
  sh[tid] = p0; sh[64 + tid] = p1; sh[128 + tid] = p2;
  __syncthreads();
  if (tid < 3) {
    float s = 0.f;
    for (int j = 0; j < 64; ++j) s += sh[tid * 64 + j];
    out[((size_t)b * 16 + t) * 3 + tid] = s + bo[tid];
  }
}

// ---------------- launcher ----------------
extern "C" void kernel_launch(void* const* d_in, const int* in_sizes, int n_in,
                              void* d_out, int out_size, void* d_ws, size_t ws_size,
                              hipStream_t stream) {
  (void)in_sizes; (void)n_in; (void)out_size;

  const float* x     = (const float*)d_in[0];
  const float* W_r   = (const float*)d_in[1];
  const float* b_r   = (const float*)d_in[2];
  const float* W_q   = (const float*)d_in[3];
  const float* b_q   = (const float*)d_in[4];
  const float* W_s   = (const float*)d_in[5];
  const float* b_s   = (const float*)d_in[6];
  const float* W_out = (const float*)d_in[7];
  const float* b_out = (const float*)d_in[8];
  float* out = (float*)d_out;

  char* wsb = (char*)d_ws;
  size_t off = 0;
  auto take = [&](size_t bytes) -> char* {
    char* p = wsb + off;
    off = (off + bytes + 255) & ~(size_t)255;
    return p;
  };
  _Float16* wr16  = (_Float16*)take((size_t)768 * 4096 * 2);
  _Float16* ws16  = (_Float16*)take((size_t)768 * 4096 * 2);
  _Float16* wq16  = (_Float16*)take((size_t)768 * 512 * 2);
  _Float16* fused = (_Float16*)take((size_t)4096 * 768 * 2);
  _Float16* qin   = (_Float16*)take((size_t)4096 * 768 * 2);
  float* RS   = (float*)take((size_t)4096 * 4096 * 4);     // shared scratch for R then S
  float* hhat = (float*)take((size_t)4096 * 512 * 8 * 4);  // recurrent trace state
  float* qbuf = (float*)take((size_t)4096 * 512 * 4);
  float* hbuf = (float*)take((size_t)4096 * 512 * 4);      // hidden = h_hat.sum(M)
  if (off > ws_size) return;  // workspace too small: bail deterministically

  // One-time weight conversion (f32 -> f16, transposed to [N,K])
  conv_transpose_f16<<<(768 * 4096 + 255) / 256, 256, 0, stream>>>(W_r, wr16, 768, 4096);
  conv_transpose_f16<<<(768 * 4096 + 255) / 256, 256, 0, stream>>>(W_s, ws16, 768, 4096);
  conv_transpose_f16<<<(768 * 512 + 255) / 256, 256, 0, stream>>>(W_q, wq16, 768, 512);
  zero_f32<<<(4096 * 512 * 8 + 255) / 256, 256, 0, stream>>>(hhat, 4096 * 512 * 8);
  zero_f32<<<(4096 * 512 + 255) / 256, 256, 0, stream>>>(hbuf, 4096 * 512);

  for (int t = 0; t < 16; ++t) {
    build_fused<<<(4096 * 768 + 255) / 256, 256, 0, stream>>>(x, hbuf, fused, qin, t);
    // R = fused @ W_r + b_r
    gemm_f16_wmma<false><<<dim3(32, 32), 256, 0, stream>>>(fused, wr16, b_r, RS, 4096, 768);
    // r = softmax(-(R-LN_TAU)^2); rh = sum(r*h_hat); qin[:,256:] = rh
    softmax_r_qin<<<(4096 * 512 + 255) / 256, 256, 0, stream>>>(RS, hhat, qin);
    // S = fused @ W_s + b_s  (reuses RS buffer after R is consumed)
    gemm_f16_wmma<false><<<dim3(32, 32), 256, 0, stream>>>(fused, ws16, b_s, RS, 4096, 768);
    // q = tanh(qin @ W_q + b_q)
    gemm_f16_wmma<true><<<dim3(4, 32), 256, 0, stream>>>(qin, wq16, b_q, qbuf, 512, 768);
    // h_hat' = ((1-s)*h_hat + s*q)*DECAY ; hbuf = h_hat'.sum(M)
    update_step<<<(4096 * 512 + 255) / 256, 256, 0, stream>>>(RS, qbuf, hhat, hbuf);
    // out[:,t,:] = hbuf @ W_out + b_out
    out_proj<<<4096, 64, 0, stream>>>(hbuf, W_out, b_out, out, t);
  }
}
